// FCNSubNet_66348654789118
// MI455X (gfx1250) — matrix operand order repaired and checked by reading, not compiled, not run on hardware
//
#include <hip/hip_runtime.h>

// Deformable-conv 3-layer stack for MI455X (gfx1250, wave32, WMMA).
// GEMM part (einsum bckhw,ock->bohw; M=256,N=36864,K=2304; ~130 GFLOP total)
// runs on v_wmma_f32_16x16x32_f16. Sampled activations are staged as an
// f16 [64 pix][2304 ck] tile in LDS (289 KB -- uses CDNA5's 320KB WGP LDS).
// Offset-conv weights are staged into LDS with GLOBAL_LOAD_ASYNC_TO_LDS_B128
// (ASYNCcnt-tracked, bypasses VGPRs).

#define B_    4
#define C_    256
#define H_    96
#define W_    96
#define HW    9216           // H_*W_
#define K_    9
#define CK    2304           // C_*K_
#define NPIX  (B_ * HW)      // 36864
#define TILE_N 64
#define SROW  2312           // padded LDS row length in halves (2304 + 8)
#define SMEM_GEMM_BYTES (TILE_N * SROW * 2)  // 295,936 B  (< 320KB WGP LDS)
#define SMEM_OFF_BYTES  (18 * CK * 4)        // 165,888 B  (< 320KB WGP LDS)

typedef _Float16 v16h __attribute__((ext_vector_type(16)));
typedef _Float16 v8h  __attribute__((ext_vector_type(8)));
typedef float    v8f  __attribute__((ext_vector_type(8)));

// Contiguous 16 halves -> v16h (unused; B fragments are loaded with load_afrag)
static __device__ __forceinline__ v16h load_v16h(const _Float16* p) {
    v8h lo = *(const v8h*)(p);
    v8h hi = *(const v8h*)(p + 8);
    v16h r;
#pragma unroll
    for (int i = 0; i < 8; ++i) { r[i] = lo[i]; r[i + 8] = hi[i]; }
    return r;
}

// A fragment per ISA layout: halves[0..7] = K at +0, halves[8..15] = K at +16
static __device__ __forceinline__ v16h load_afrag(const _Float16* p) {
    v8h lo = *(const v8h*)(p);
    v8h hi = *(const v8h*)(p + 16);
    v16h r;
#pragma unroll
    for (int i = 0; i < 8; ++i) { r[i] = lo[i]; r[i + 8] = hi[i]; }
    return r;
}

// ---------------- weight f32 -> f16 conversion (one-time) ----------------
__global__ void wcvt_kernel(const float* __restrict__ w, _Float16* __restrict__ w16, int n) {
    int i = blockIdx.x * blockDim.x + threadIdx.x;
    if (i < n) w16[i] = (_Float16)w[i];
}

// ---------------- offset conv: 3x3, C=256 -> 18 channels ----------------
// Weights (18*2304 f32 = 162 KB) staged in LDS via async global->LDS DMA;
// one thread per output pixel.
__global__ void offset_conv_kernel(const float* __restrict__ x,
                                   const float* __restrict__ w_off,
                                   const float* __restrict__ b_off,
                                   float* __restrict__ off) {
    extern __shared__ char dyn_smem[];
    float* lw = (float*)dyn_smem;
    const int tid = threadIdx.x;

    // Async-copy 18*CK floats (10368 x 16B) into LDS; no VGPR data movement.
    // Generic pointers to LDS carry the LDS byte offset in addr[31:0].
    for (int i = tid; i < (18 * CK) / 4; i += 256) {
        const unsigned lds_addr = (unsigned)(size_t)(const void*)(lw + i * 4);
        const float*   gptr     = w_off + i * 4;
        asm volatile("global_load_async_to_lds_b128 %0, %1, off"
                     :: "v"(lds_addr), "v"(gptr) : "memory");
    }
    asm volatile("s_wait_asynccnt 0x0" ::: "memory");
    __syncthreads();

    const int g  = blockIdx.x * 256 + tid;   // NPIX is a multiple of 256
    const int b  = g / HW;
    const int hw = g % HW;
    const int h  = hw / W_;
    const int w  = hw % W_;

    float acc[18];
#pragma unroll
    for (int oc = 0; oc < 18; ++oc) acc[oc] = b_off[oc];

    const float* xb = x + (size_t)b * (C_ * HW);
#pragma unroll 1
    for (int tap = 0; tap < 9; ++tap) {
        const int yy = h - 1 + tap / 3;
        const int xx = w - 1 + tap % 3;
        if (yy < 0 || yy >= H_ || xx < 0 || xx >= W_) continue;
        const float* xp = xb + yy * W_ + xx;
        const float* wp = lw + tap;
#pragma unroll 1
        for (int c = 0; c < C_; ++c) {
            const float xv = xp[(size_t)c * HW];
#pragma unroll
            for (int oc = 0; oc < 18; ++oc)
                acc[oc] = fmaf(xv, wp[oc * CK + c * 9], acc[oc]);
        }
    }
#pragma unroll
    for (int oc = 0; oc < 18; ++oc)
        off[((size_t)b * 18 + oc) * HW + hw] = acc[oc];
}

// ---------------- deformable gather + WMMA GEMM + ReLU ----------------
// One workgroup = 64 pixels. Phase A: bilinear-sample samp[pix][ck] (f16) into
// LDS. Phase B: 8 waves, each computes 32(O) x 64(N) via 2x4 WMMA tiles.
__global__ void deform_gemm_kernel(const float* __restrict__ x,
                                   const float* __restrict__ off,
                                   const _Float16* __restrict__ w16,
                                   float* __restrict__ out) {
    extern __shared__ char dyn_smem[];
    _Float16* samp = (_Float16*)dyn_smem;
    const int tid = threadIdx.x;
    const int pixBase = blockIdx.x * TILE_N;

    // ---- Phase A: gather 64 pixels x 9 taps x 256 channels into LDS (f16)
    for (int pair = tid; pair < TILE_N * K_; pair += 256) {
        const int pix = pair & (TILE_N - 1);
        const int k   = pair >> 6;
        const int g   = pixBase + pix;
        const int b   = g / HW;
        const int hw  = g % HW;
        const int h   = hw / W_;
        const int w   = hw % W_;

        const float offy = off[((size_t)b * 18 + 2 * k)     * HW + hw];
        const float offx = off[((size_t)b * 18 + 2 * k + 1) * HW + hw];
        const float py = (float)(h - 1 + k / 3) + offy;
        const float px = (float)(w - 1 + k % 3) + offx;
        const float y0f = floorf(py), x0f = floorf(px);
        const float wy = py - y0f,  wx = px - x0f;
        const int iy0 = (int)y0f, ix0 = (int)x0f;
        const int iy1 = iy0 + 1,  ix1 = ix0 + 1;
        const float vy0 = (iy0 >= 0 && iy0 < H_) ? 1.f : 0.f;
        const float vy1 = (iy1 >= 0 && iy1 < H_) ? 1.f : 0.f;
        const float vx0 = (ix0 >= 0 && ix0 < W_) ? 1.f : 0.f;
        const float vx1 = (ix1 >= 0 && ix1 < W_) ? 1.f : 0.f;
        const int cy0 = iy0 < 0 ? 0 : (iy0 > H_ - 1 ? H_ - 1 : iy0);
        const int cy1 = iy1 < 0 ? 0 : (iy1 > H_ - 1 ? H_ - 1 : iy1);
        const int cx0 = ix0 < 0 ? 0 : (ix0 > W_ - 1 ? W_ - 1 : ix0);
        const int cx1 = ix1 < 0 ? 0 : (ix1 > W_ - 1 ? W_ - 1 : ix1);
        const float w00 = (1.f - wy) * (1.f - wx) * vy0 * vx0;
        const float w01 = (1.f - wy) * wx         * vy0 * vx1;
        const float w10 = wy         * (1.f - wx) * vy1 * vx0;
        const float w11 = wy         * wx         * vy1 * vx1;
        const int o00 = cy0 * W_ + cx0, o01 = cy0 * W_ + cx1;
        const int o10 = cy1 * W_ + cx0, o11 = cy1 * W_ + cx1;

        const float* xb = x + (size_t)b * (C_ * HW);
        _Float16* dst = samp + pix * SROW + k;
#pragma unroll 1
        for (int c = 0; c < C_; ++c) {
            const float* xc = xb + (size_t)c * HW;
            const float v = w00 * xc[o00] + w01 * xc[o01] +
                            w10 * xc[o10] + w11 * xc[o11];
            dst[c * 9] = (_Float16)v;
        }
    }
    __syncthreads();

    // ---- Phase B: WMMA GEMM. wave -> O-tiles {2*wave, 2*wave+1}, all 4 N-tiles.
    const int lane   = tid & 31;
    const int waveId = tid >> 5;
    const int col    = lane & 15;
    const int hiHalf = lane >> 4;

    v8f acc[2][4] = {};
    const _Float16* aBase0 = w16 + (size_t)(waveId * 32 + col) * CK;
    const _Float16* aBase1 = aBase0 + (size_t)16 * CK;

#pragma unroll 1
    for (int kk = 0; kk < CK / 32; ++kk) {
        const int kcol = kk * 32;
        const v16h a0 = load_afrag(aBase0 + kcol + hiHalf * 8);
        const v16h a1 = load_afrag(aBase1 + kcol + hiHalf * 8);
        __builtin_prefetch(aBase0 + kcol + 64, 0, 0);  // global_prefetch next A
        __builtin_prefetch(aBase1 + kcol + 64, 0, 0);

        v16h bf[4];
#pragma unroll
        for (int nt = 0; nt < 4; ++nt)
            bf[nt] = load_afrag(samp + (nt * 16 + col) * SROW + kcol + hiHalf * 8);

#pragma unroll
        for (int nt = 0; nt < 4; ++nt) {
            acc[0][nt] = __builtin_amdgcn_wmma_f32_16x16x32_f16(
                false, a0, false, bf[nt], (short)0, acc[0][nt], false, false);
            acc[1][nt] = __builtin_amdgcn_wmma_f32_16x16x32_f16(
                false, a1, false, bf[nt], (short)0, acc[1][nt], false, false);
        }
    }

    // ---- Epilogue: ReLU + scatter to NCHW
#pragma unroll
    for (int nt = 0; nt < 4; ++nt) {
        const int g  = pixBase + nt * 16 + col;
        const int b  = g / HW;
        const int hw = g % HW;
        float* ob = out + (size_t)b * (C_ * HW) + hw;
#pragma unroll
        for (int ot = 0; ot < 2; ++ot) {
            const int oBase = (waveId * 2 + ot) * 16;
#pragma unroll
            for (int r = 0; r < 8; ++r) {
                const int m = r + hiHalf * 8;        // C/D layout: M = r + 8*(lane>=16)
                float v = acc[ot][nt][r];
                v = v > 0.f ? v : 0.f;
                ob[(size_t)(oBase + m) * HW] = v;
            }
        }
    }
}

// ---------------------------------------------------------------------------
extern "C" void kernel_launch(void* const* d_in, const int* in_sizes, int n_in,
                              void* d_out, int out_size, void* d_ws, size_t ws_size,
                              hipStream_t stream) {
    const float* x = (const float*)d_in[0];
    const float* w_off[3] = {(const float*)d_in[1], (const float*)d_in[4], (const float*)d_in[7]};
    const float* b_off[3] = {(const float*)d_in[2], (const float*)d_in[5], (const float*)d_in[8]};
    const float* w[3]     = {(const float*)d_in[3], (const float*)d_in[6], (const float*)d_in[9]};

    // Workspace carve-up (~82 MB total):
    //   3x f16 weights (1.18 MB each) + offset map (2.65 MB) + 2x activations (37.7 MB each)
    char*  ws = (char*)d_ws;
    size_t p  = 0;
    auto alloc = [&](size_t bytes) -> char* {
        p = (p + 255) & ~(size_t)255;
        char* r = ws + p;
        p += bytes;
        return r;
    };
    _Float16* w16[3];
    for (int i = 0; i < 3; ++i) w16[i] = (_Float16*)alloc((size_t)C_ * CK * sizeof(_Float16));
    float* offbuf = (float*)alloc((size_t)B_ * 18 * HW * sizeof(float));
    float* y0     = (float*)alloc((size_t)B_ * C_ * HW * sizeof(float));
    float* y1     = (float*)alloc((size_t)B_ * C_ * HW * sizeof(float));

    for (int i = 0; i < 3; ++i) {
        const int n = C_ * CK;
        wcvt_kernel<<<(n + 255) / 256, 256, 0, stream>>>(w[i], w16[i], n);
    }

    const float* cur = x;
    float* outs[3] = {y0, y1, (float*)d_out};
    for (int i = 0; i < 3; ++i) {
        offset_conv_kernel<<<NPIX / 256, 256, SMEM_OFF_BYTES, stream>>>(
            cur, w_off[i], b_off[i], offbuf);
        deform_gemm_kernel<<<NPIX / TILE_N, 256, SMEM_GEMM_BYTES, stream>>>(
            cur, offbuf, w16[i], outs[i]);
        cur = outs[i];
    }
}
